// SelfAttention_163208757571
// MI455X (gfx1250) — matrix operand order repaired and checked by reading, not compiled, not run on hardware
//
#include <hip/hip_runtime.h>
#include <hip/hip_bf16.h>

// ---------- WMMA types / helpers (CDNA5 gfx1250, wave32) ----------
typedef __attribute__((ext_vector_type(16))) __bf16 v16bf;
typedef __attribute__((ext_vector_type(8)))  float  v8f;

union FragB16 { v16bf v; unsigned int u[8]; unsigned short h[16]; };
union FragF32 { v8f   v; float f[8]; };

__device__ __forceinline__ unsigned short f2bf_bits(float f) {
    union { __bf16 b; unsigned short s; } c; c.b = (__bf16)f; return c.s;
}
__device__ __forceinline__ unsigned pack2(float lo, float hi) {
    union { __bf16 b[2]; unsigned u; } c;
    c.b[0] = (__bf16)lo; c.b[1] = (__bf16)hi; return c.u;
}

// CDNA5 async global->LDS copy (ASYNCcnt-tracked), GVS addressing:
//   lds[vdst_vgpr] <- mem[saddr64 + voffset32]
__device__ __forceinline__ void async_load_b128(unsigned lds_off, unsigned voff,
                                                unsigned long long sbase) {
    asm volatile("global_load_async_to_lds_b128 %0, %1, %2"
                 :: "v"(lds_off), "v"(voff), "s"(sbase) : "memory");
}
__device__ __forceinline__ void wait_async0() {
#if __has_builtin(__builtin_amdgcn_s_wait_asynccnt)
    __builtin_amdgcn_s_wait_asynccnt(0);
#else
    asm volatile("s_wait_asynccnt 0x0" ::: "memory");
#endif
}

#define N_PIX 4096
#define C_IN  256
#define DQK   32
#define NT    64    // key tile
#define MT    32    // query rows per workgroup
#define N_OB  20    // 320 output channels / 16

// ============================================================================
// Stage 0: pre-pack W = [Wq;Wk;Wv] (f32) into bf16 B-fragment order:
//   Wc[ob][ks][lane][v] = pack2(W[o][c], W[o][c+1]),
//   o = oBase(ob) + (lane&15), c = ks*32 + (lane>>4)*16 + v*2.
// One-time, 160 KB -> proj kernel B-frags become two b128 loads.
// ============================================================================
__global__ __launch_bounds__(256) void pack_w_kernel(
    const float* __restrict__ Wq, const float* __restrict__ Wk,
    const float* __restrict__ Wv, unsigned* __restrict__ Wc)
{
    const int t = blockIdx.x * 256 + threadIdx.x;      // 20*8*32 = 5120 threads
    if (t >= N_OB * 8 * 32) return;
    const int lane = t & 31;
    const int ks   = (t >> 5) & 7;
    const int ob   = t >> 8;
    const float* W; int oBase;
    if (ob < 2)      { W = Wq; oBase = ob * 16; }
    else if (ob < 4) { W = Wk; oBase = (ob - 2) * 16; }
    else             { W = Wv; oBase = (ob - 4) * 16; }
    const int o     = oBase + (lane & 15);
    const int cBase = ks * 32 + (lane >> 4) * 16;
    const float* src = W + (size_t)o * C_IN + cBase;
    unsigned* dst = Wc + (size_t)t * 8;
    #pragma unroll
    for (int v = 0; v < 8; ++v)
        dst[v] = pack2(src[2 * v], src[2 * v + 1]);
}

// ============================================================================
// Stage 1: fused q/k/v projection (1x1 conv == GEMM), f32 in -> bf16 out.
//   qg,kg = [B][N][32]  (channel-contiguous rows: A-frag friendly)
//   vg    = [B][256][N] (key-contiguous rows: transposed LDS staging friendly)
// Grid: B * (N/64) blocks, 256 threads (8 waves).
// ============================================================================
__global__ __launch_bounds__(256) void proj_kernel(
    const float* __restrict__ x,
    const float* __restrict__ bq, const float* __restrict__ bk,
    const float* __restrict__ bv,
    const unsigned* __restrict__ Wc,
    unsigned short* __restrict__ qg, unsigned short* __restrict__ kg,
    unsigned short* __restrict__ vg)
{
    __shared__ __align__(16) unsigned short xs[64 * C_IN];   // [pix][c], 32 KB

    const int tid  = threadIdx.x;
    const int wave = tid >> 5;
    const int lane = tid & 31;
    const int blk  = blockIdx.x;
    const int b    = blk >> 6;          // 64 pixel-tiles per batch
    const int n0   = (blk & 63) * 64;

    // Stage x tile transposed into LDS as packed bf16 pairs.
    // Thread t owns channel pair c=(t&127)*2 and pixel half ph=t>>7.
    {
        const int cpair = (tid & 127) * 2;
        const int ph    = tid >> 7;
        const float* r0 = x + ((size_t)b * C_IN + cpair)     * N_PIX + n0 + ph * 32;
        const float* r1 = x + ((size_t)b * C_IN + cpair + 1) * N_PIX + n0 + ph * 32;
        #pragma unroll
        for (int i = 0; i < 32; i += 4) {
            float4 a4 = *(const float4*)(r0 + i);
            float4 b4 = *(const float4*)(r1 + i);
            const int p = ph * 32 + i;
            *(unsigned*)(xs + (p + 0) * C_IN + cpair) = pack2(a4.x, b4.x);
            *(unsigned*)(xs + (p + 1) * C_IN + cpair) = pack2(a4.y, b4.y);
            *(unsigned*)(xs + (p + 2) * C_IN + cpair) = pack2(a4.z, b4.z);
            *(unsigned*)(xs + (p + 3) * C_IN + cpair) = pack2(a4.w, b4.w);
        }
    }
    __syncthreads();

    const int mb     = wave & 3;            // which 16 pixels
    const int obBase = (wave >> 2) * 10;    // out-blocks 0..9 or 10..19
    const int half   = lane >> 4;
    const int l15    = lane & 15;
    const int pixOff = mb * 16;
    const unsigned short* arowBase = xs + (pixOff + l15) * C_IN + half * 16;

    for (int obi = 0; obi < 10; ++obi) {
        const int ob = __builtin_amdgcn_readfirstlane(obBase + obi);  // scalar

        v8f acc = {};
        for (int ks = 0; ks < 8; ++ks) {          // K loop over C=256 in 32-chunks
            const int c0 = ks * 32;
            FragB16 a, bf;
            // A-frag: two contiguous 16B runs per lane
            *(uint4*)&a.u[0] = *(const uint4*)(arowBase + c0);
        *(uint4*)&a.u[4] = *(const uint4*)(arowBase + c0 + 8);
            // B-frag: pre-packed bf16, two b128 global loads
            const unsigned* wrow = Wc + (((size_t)(ob * 8 + ks) * 32 + lane) << 3);
            *(uint4*)&bf.u[0] = *(const uint4*)(wrow);
            *(uint4*)&bf.u[4] = *(const uint4*)(wrow + 4);
            acc = __builtin_amdgcn_wmma_f32_16x16x32_bf16(
                      false, a.v, false, bf.v, (short)0, acc, false, false);
        }

        // Store D (lane: n = oBase+l15, rows m = r+8*half). ob is scalar.
        FragF32 d; d.v = acc;
        if (ob < 4) {
            const int region0 = (ob < 2);
            const int oBase = (region0 ? ob : (ob - 2)) * 16;
            const float bo = (region0 ? bq : bk)[oBase + l15];
            unsigned short* dst = (region0 ? qg : kg) + (size_t)b * N_PIX * DQK;
            #pragma unroll
            for (int r = 0; r < 8; ++r) {
                const int pix = n0 + pixOff + r + 8 * half;
                dst[(size_t)pix * DQK + oBase + l15] = f2bf_bits(d.f[r] + bo);
            }
        } else {
            const int oBase = (ob - 4) * 16;
            const float bo = bv[oBase + l15];
            // v region: r indexes consecutive pixels -> packed dword stores
            unsigned short* dst = vg + ((size_t)b * C_IN + oBase + l15) * N_PIX
                                     + n0 + pixOff + 8 * half;
            #pragma unroll
            for (int r = 0; r < 8; r += 2)
                *(unsigned*)(dst + r) = pack2(d.f[r] + bo, d.f[r + 1] + bo);
        }
    }
}

// ============================================================================
// Stage 2: flash attention. Grid: B * (N/32) blocks, 256 threads (8 waves).
// Wave w: mb = w&1 (16 of 32 query rows), vc = w>>1 (64 of 256 V channels).
// K/V tiles staged via GLOBAL_LOAD_ASYNC_TO_LDS_B128 (ASYNCcnt).
// ============================================================================
__global__ __launch_bounds__(256) void attn_kernel(
    const unsigned short* __restrict__ qg, const unsigned short* __restrict__ kg,
    const unsigned short* __restrict__ vg, const float* __restrict__ x,
    const float* __restrict__ gamma, float* __restrict__ out)
{
    __shared__ __align__(16) unsigned short Kt[NT * DQK];   // [j][c]    4 KB
    __shared__ __align__(16) unsigned short Vt[C_IN * NT];  // [c][j]   32 KB
    __shared__ __align__(16) unsigned short Pl[MT * NT];    // [m][j]    4 KB

    const int tid  = threadIdx.x;
    const int lane = tid & 31;
    const int wave = tid >> 5;
    const int mb   = wave & 1;
    const int vc   = wave >> 1;
    const int half = lane >> 4;
    const int l15  = lane & 15;
    const int blk  = blockIdx.x;
    const int b    = blk >> 7;                  // 128 query-tiles per batch
    const int q0   = (blk & 127) * MT;
    const int vcScalar = __builtin_amdgcn_readfirstlane(vc);

    const unsigned long long kBase = (unsigned long long)(uintptr_t)(kg + (size_t)b * N_PIX * DQK);
    const unsigned long long vBase = (unsigned long long)(uintptr_t)(vg + (size_t)b * C_IN * N_PIX);

    // Q A-fragment, loaded once (d=32 == one WMMA K-step): two b128 loads.
    FragB16 qa;
    {
        const int pix = q0 + mb * 16 + l15;
        const unsigned short* qrow = qg + ((size_t)b * N_PIX + pix) * DQK + half * 8;
        *(uint4*)&qa.u[0] = *(const uint4*)(qrow);
        *(uint4*)&qa.u[4] = *(const uint4*)(qrow + 16);
    }

    v8f o0 = {}, o1 = {}, o2 = {}, o3 = {};     // O slice: 16 rows x 64 channels
    float mrow[8], lrow[8];
    #pragma unroll
    for (int r = 0; r < 8; ++r) { mrow[r] = -1e30f; lrow[r] = 0.0f; }

    for (int jt = 0; jt < N_PIX / NT; ++jt) {
        const int j0 = jt * NT;
        __syncthreads();    // previous-iteration LDS consumers done

        // --- async stage K tile [64][32] (one b128/thread)
        {
            const int j = tid >> 2, c0 = (tid & 3) * 8;
            async_load_b128((unsigned)(uintptr_t)&Kt[j * DQK + c0],
                            (unsigned)(((j0 + j) * DQK + c0) * 2), kBase);
        }
        // --- async stage V^T tile [256][64] (8x b128/thread, row c = tid)
        #pragma unroll
        for (int i = 0; i < 8; ++i) {
            async_load_b128((unsigned)(uintptr_t)&Vt[tid * NT + i * 8],
                            (unsigned)((tid * N_PIX + j0 + i * 8) * 2), vBase);
        }
        // prefetch next V tile row into caches while this tile computes
        if (jt + 1 < N_PIX / NT)
            __builtin_prefetch(vg + (size_t)b * C_IN * N_PIX + (size_t)tid * N_PIX + j0 + NT, 0, 0);
        wait_async0();
        __syncthreads();

        // --- S = Q * K^T  (16 x 64, four 16x16 WMMAs, K=32 in one step)
        FragF32 sf[4];
        #pragma unroll
        for (int nb = 0; nb < 4; ++nb) {
            FragB16 kb;
            const unsigned short* krow = Kt + (nb * 16 + l15) * DQK + half * 8;
            *(uint4*)&kb.u[0] = *(const uint4*)(krow);
            *(uint4*)&kb.u[4] = *(const uint4*)(krow + 16);
            v8f z = {};
            sf[nb].v = __builtin_amdgcn_wmma_f32_16x16x32_bf16(
                           false, qa.v, false, kb.v, (short)0, z, false, false);
        }

        // --- online softmax (rows m = r + 8*half; columns across 16 lanes)
        #pragma unroll
        for (int r = 0; r < 8; ++r) {
            float tm = fmaxf(fmaxf(sf[0].f[r], sf[1].f[r]),
                             fmaxf(sf[2].f[r], sf[3].f[r]));
            tm = fmaxf(tm, __shfl_xor(tm, 1, 32));
            tm = fmaxf(tm, __shfl_xor(tm, 2, 32));
            tm = fmaxf(tm, __shfl_xor(tm, 4, 32));
            tm = fmaxf(tm, __shfl_xor(tm, 8, 32));
            const float mnew  = fmaxf(mrow[r], tm);
            const float alpha = __expf(mrow[r] - mnew);
            mrow[r] = mnew;
            float rs = 0.0f;
            #pragma unroll
            for (int nb = 0; nb < 4; ++nb) {
                const float p = __expf(sf[nb].f[r] - mnew);
                sf[nb].f[r] = p;
                rs += p;
            }
            rs += __shfl_xor(rs, 1, 32);
            rs += __shfl_xor(rs, 2, 32);
            rs += __shfl_xor(rs, 4, 32);
            rs += __shfl_xor(rs, 8, 32);
            lrow[r] = lrow[r] * alpha + rs;
            o0[r] *= alpha; o1[r] *= alpha; o2[r] *= alpha; o3[r] *= alpha;
        }

        // --- publish P (bf16) to LDS; one wave per row-half suffices (scalar branch)
        if (vcScalar == 0) {
            #pragma unroll
            for (int nb = 0; nb < 4; ++nb)
                #pragma unroll
                for (int r = 0; r < 8; ++r)
                    Pl[(mb * 16 + r + 8 * half) * NT + nb * 16 + l15] =
                        f2bf_bits(sf[nb].f[r]);
        }
        __syncthreads();

        // --- O += P * V   (contraction over 64 keys = 2 WMMA K-steps)
        #pragma unroll
        for (int ksv = 0; ksv < 2; ++ksv) {
            FragB16 pa;
            const unsigned short* prow = Pl + (mb * 16 + l15) * NT + ksv * 32 + half * 8;
            *(uint4*)&pa.u[0] = *(const uint4*)(prow);
            *(uint4*)&pa.u[4] = *(const uint4*)(prow + 16);
            #pragma unroll
            for (int nb = 0; nb < 4; ++nb) {
                FragB16 vb;
                const unsigned short* vrow =
                    Vt + (vc * 64 + nb * 16 + l15) * NT + ksv * 32 + half * 8;
                *(uint4*)&vb.u[0] = *(const uint4*)(vrow);
                *(uint4*)&vb.u[4] = *(const uint4*)(vrow + 16);
                v8f* op = (nb == 0) ? &o0 : (nb == 1) ? &o1 : (nb == 2) ? &o2 : &o3;
                *op = __builtin_amdgcn_wmma_f32_16x16x32_bf16(
                          false, pa.v, false, vb.v, (short)0, *op, false, false);
            }
        }
    }

    // --- epilogue: out = gamma * (O / l) + x  (r = consecutive pixels -> float4)
    const float g = gamma[0];
    FragF32 of[4]; of[0].v = o0; of[1].v = o1; of[2].v = o2; of[3].v = o3;
    float inv[8];
    #pragma unroll
    for (int r = 0; r < 8; ++r) inv[r] = 1.0f / lrow[r];
    const int pixBase = q0 + mb * 16 + 8 * half;
    #pragma unroll
    for (int nb = 0; nb < 4; ++nb) {
        const int ch = vc * 64 + nb * 16 + l15;
        const size_t base = ((size_t)b * C_IN + ch) * N_PIX + pixBase;
        float4 x0 = *(const float4*)(x + base);
        float4 x1 = *(const float4*)(x + base + 4);
        float4 r0, r1;
        r0.x = g * of[nb].f[0] * inv[0] + x0.x;
        r0.y = g * of[nb].f[1] * inv[1] + x0.y;
        r0.z = g * of[nb].f[2] * inv[2] + x0.z;
        r0.w = g * of[nb].f[3] * inv[3] + x0.w;
        r1.x = g * of[nb].f[4] * inv[4] + x1.x;
        r1.y = g * of[nb].f[5] * inv[5] + x1.y;
        r1.z = g * of[nb].f[6] * inv[6] + x1.z;
        r1.w = g * of[nb].f[7] * inv[7] + x1.w;
        *(float4*)(out + base)     = r0;
        *(float4*)(out + base + 4) = r1;
    }
}

// ============================================================================
extern "C" void kernel_launch(void* const* d_in, const int* in_sizes, int n_in,
                              void* d_out, int out_size, void* d_ws, size_t ws_size,
                              hipStream_t stream) {
    const float* x     = (const float*)d_in[0];
    const float* Wq    = (const float*)d_in[1];
    const float* bq    = (const float*)d_in[2];
    const float* Wk    = (const float*)d_in[3];
    const float* bk    = (const float*)d_in[4];
    const float* Wv    = (const float*)d_in[5];
    const float* bv    = (const float*)d_in[6];
    const float* gamma = (const float*)d_in[7];
    float* out = (float*)d_out;

    // Workspace partition: q 1MB | k 1MB | v 8MB | Wc 160KB
    unsigned short* qg = (unsigned short*)d_ws;
    unsigned short* kg = qg + (size_t)4 * N_PIX * DQK;
    unsigned short* vg = kg + (size_t)4 * N_PIX * DQK;
    unsigned*       Wc = (unsigned*)(vg + (size_t)4 * C_IN * N_PIX);

    hipLaunchKernelGGL(pack_w_kernel, dim3(N_OB), dim3(256), 0, stream,
                       Wq, Wk, Wv, Wc);
    hipLaunchKernelGGL(proj_kernel, dim3(4 * (N_PIX / 64)), dim3(256), 0, stream,
                       x, bq, bk, bv, Wc, qg, kg, vg);
    hipLaunchKernelGGL(attn_kernel, dim3(4 * (N_PIX / MT)), dim3(256), 0, stream,
                       qg, kg, vg, x, gamma, out);
}
